// TCN_GCN_unit_40303973105742
// MI455X (gfx1250) — compile-verified, matrix operand order checked by reading, not run on hardware
//
#include <hip/hip_runtime.h>
#include <hip/hip_bf16.h>
#include <stdint.h>

// ---------------------------------------------------------------------------
// MI455X / gfx1250 implementation of the TCN_GCN unit.
// Wave32, WMMA f32<-f16 16x16x32 for all channel-mixing GEMMs.
// Async global->LDS staging (ASYNCcnt) when the toolchain exposes it.
// ---------------------------------------------------------------------------

typedef __attribute__((ext_vector_type(16))) _Float16 v16h;
typedef __attribute__((ext_vector_type(8)))  float    v8f;
typedef int v4i_t __attribute__((vector_size(16)));

#define NB    64
#define CCH   64
#define TTT   300
#define VVV   25
#define NTV   (NB*TTT*VVV)       /* 480000 */
#define LSTR  40                 /* LDS row stride in halves: 80B, 16B-aligned */

// --- async global->LDS (gfx1250 GLOBAL_LOAD_ASYNC_TO_LDS_B128) -------------
#if defined(__gfx1250__) && __has_builtin(__builtin_amdgcn_global_load_async_to_lds_b128)
#define ASYNC_LDS 1
#else
#define ASYNC_LDS 0
#endif

__device__ __forceinline__ void copy16_g2l(_Float16* lds_dst, const _Float16* gsrc) {
#if ASYNC_LDS
  __builtin_amdgcn_global_load_async_to_lds_b128(
      (__attribute__((address_space(1))) v4i_t*)gsrc,
      (__attribute__((address_space(3))) v4i_t*)lds_dst, 0, 0);
#else
  *(uint4*)lds_dst = *(const uint4*)gsrc;
#endif
}

__device__ __forceinline__ void async_wait_all() {
#if ASYNC_LDS
#if __has_builtin(__builtin_amdgcn_s_wait_asynccnt)
  __builtin_amdgcn_s_wait_asynccnt(0);
#else
  asm volatile("s_wait_asynccnt 0x0" ::: "memory");
#endif
#endif
}

// ---------------------------------------------------------------------------
// prep kernels (tiny)
// ---------------------------------------------------------------------------
__global__ void k_prep_bn(const float* __restrict__ g, const float* __restrict__ b,
                          const float* __restrict__ m, const float* __restrict__ v,
                          float* __restrict__ s, float* __restrict__ sh, int C) {
  int i = blockIdx.x * blockDim.x + threadIdx.x;
  if (i < C) {
    float sc = g[i] * rsqrtf(v[i] + 1e-5f);
    s[i]  = sc;
    sh[i] = b[i] - m[i] * sc;
  }
}

__global__ void k_prep_wab(const float* __restrict__ Wa, const float* __restrict__ ba,
                           const float* __restrict__ Wb, const float* __restrict__ bb,
                           _Float16* __restrict__ wab, float* __restrict__ biasab) {
  int idx = blockIdx.x * blockDim.x + threadIdx.x;
  if (idx < 64 * 96) {
    int k = idx / 96, j = idx % 96;
    float w;
    if (j < 48)      w = Wa[(j / 16) * 1024 + (j % 16) * 64 + k];
    else { int jj = j - 48; w = Wb[(jj / 16) * 1024 + (jj % 16) * 64 + k]; }
    wab[k * 96 + j] = (_Float16)w;
  }
  if (idx < 96) biasab[idx] = (idx < 48) ? ba[idx] : bb[idx - 48];
}

__global__ void k_prep_wd(const float* __restrict__ Wd, const float* __restrict__ bd,
                          _Float16* __restrict__ wdp, float* __restrict__ sumbd) {
  int idx = blockIdx.x * blockDim.x + threadIdx.x;
  if (idx < 64 * 192) {
    int k = idx / 192, j = idx % 192;
    int i = j / 64, co = j % 64;
    wdp[k * 192 + j] = (_Float16)Wd[i * 4096 + co * 64 + k];
  }
  if (idx < 64) sumbd[idx] = bd[idx] + bd[64 + idx] + bd[128 + idx];
}

__global__ void k_prep_w0(const float* __restrict__ W0, _Float16* __restrict__ w0h) {
  int idx = blockIdx.x * blockDim.x + threadIdx.x;
  if (idx < 64 * 32) { int k = idx / 32, j = idx % 32; w0h[k * 32 + j] = (_Float16)W0[j * 64 + k]; }
}

__global__ void k_prep_wc(const float* __restrict__ Wc, _Float16* __restrict__ wch) {
  int idx = blockIdx.x * blockDim.x + threadIdx.x;
  if (idx < 160 * 64) { int k = idx / 64, j = idx % 64; wch[k * 64 + j] = (_Float16)Wc[j * 160 + k]; }
}

__constant__ int c_tx[16] = {0,0,6,0,0,1,1,4,5,1,3,0,0,0,3,2};
__constant__ int c_ty[16] = {0,1,0,5,2,0,2,0,0,6,0,4,6,3,2,5};

__global__ void k_dctw(float* __restrict__ dctw) {
  int idx = blockIdx.x * blockDim.x + threadIdx.x;
  if (idx >= 64 * 7500) return;
  int c = idx / 7500, r = idx % 7500, t = r / 25, v = r % 25;
  int g  = c >> 2;                       /* cpart = 4 */
  int fx = c_tx[g] * 42;                 /* 300 // 7 */
  int fy = c_ty[g] * 3;                  /* 25  // 7 */
  const float PI = 3.14159265358979323846f;
  float bt = cosf(PI * fx * (t + 0.5f) / 300.f) * rsqrtf(300.f);
  if (fx) bt *= 1.41421356237f;
  float bv = cosf(PI * fy * (v + 0.5f) / 25.f) * rsqrtf(25.f);
  if (fy) bv *= 1.41421356237f;
  dctw[idx] = bt * bv;
}

// x (n,c,t,v) f32 -> xh [(n,t,v), c] f16 (GEMM-friendly rows, K contiguous)
__global__ __launch_bounds__(256) void k_xh(const float* __restrict__ x, _Float16* __restrict__ xh) {
  long idx = (long)blockIdx.x * 256 + threadIdx.x;   /* < NTV*64 */
  int  c = idx & 63;
  long s = idx >> 6;
  int  v = (int)(s % 25);
  int  t = (int)((s / 25) % 300);
  int  n = (int)(s / 7500);
  xh[idx] = (_Float16)x[(((long)(n * 64 + c)) * 300 + t) * 25 + v];
}

// ---------------------------------------------------------------------------
// Generic WMMA GEMM:  Out[M x Ncols] = A[M x K](f16,row-major,lda) *
//                                      B[K x Ncols](f16,row-major,ldb)
// Block: 128 threads (4 waves), tile BM=64 x BN=32, K step 32.
// Epilogue: optional +bias[n], optional relu(v*s[n]+sh[n]), f16 or f32 out.
// A fragment (16-bit A 16x32, ISA 7.12.2): lane L: M=L%16, half=L/16,
//   elements 0..7  -> K = 8*half + e
//   elements 8..15 -> K = 16 + 8*half + (e-8)
// => two 16B LDS loads per fragment. B assumed dual layout (N = L%16).
// A tile is staged via async global->LDS; B tile is transposed via DS stores.
// ---------------------------------------------------------------------------
__global__ __launch_bounds__(128)
void k_gemm(const _Float16* __restrict__ A, int lda,
            const _Float16* __restrict__ B, int ldb,
            int K,
            void* __restrict__ Out, int ldo, int out_f16,
            const float* __restrict__ bias,
            const float* __restrict__ bns, const float* __restrict__ bnsh) {
  __shared__ __align__(16) _Float16 la[64 * LSTR];
  __shared__ __align__(16) _Float16 lb[32 * LSTR];
  const int tid  = threadIdx.x;
  const int wave = tid >> 5, lane = tid & 31;
  const int half = lane >> 4, l16 = lane & 15;
  const long m0 = (long)blockIdx.x * 64;
  const int  n0 = blockIdx.y * 32;

  v8f acc[2] = {v8f{}, v8f{}};

  for (int k0 = 0; k0 < K; k0 += 32) {
    // stage A tile [64 x 32]: 2 x 16B async copies per thread
    {
      int q = tid * 2;
#pragma unroll
      for (int it = 0; it < 2; ++it, ++q) {
        int row = q >> 2, offh = (q & 3) << 3;
        copy16_g2l(&la[row * LSTR + offh], A + (m0 + row) * lda + k0 + offh);
      }
    }
    // stage B tile transposed into [n][k]: 1 x uint4 per thread
    {
      int kk = tid >> 2, ng = (tid & 3) << 3;
      union { uint4 u; _Float16 h[8]; } bbu;
      bbu.u = *(const uint4*)(B + (long)(k0 + kk) * ldb + n0 + ng);
#pragma unroll
      for (int j = 0; j < 8; ++j) lb[(ng + j) * LSTR + kk] = bbu.h[j];
    }
    if (k0 + 32 < K)   /* gfx1250 global_prefetch_b8 of next A tile */
      __builtin_prefetch(A + (m0 + (tid >> 1)) * lda + k0 + 32, 0, 3);
    async_wait_all();
    __syncthreads();

    union { uint4 u[2]; v16h v; } fa, fb;
    const int mr = wave * 16 + l16;
    fa.u[0] = *(const uint4*)(&la[mr * LSTR + 8 * half]);
    fa.u[1] = *(const uint4*)(&la[mr * LSTR + 16 + 8 * half]);
#pragma unroll
    for (int nt = 0; nt < 2; ++nt) {
      fb.u[0] = *(const uint4*)(&lb[(nt * 16 + l16) * LSTR + 8 * half]);
      fb.u[1] = *(const uint4*)(&lb[(nt * 16 + l16) * LSTR + 16 + 8 * half]);
      acc[nt] = __builtin_amdgcn_wmma_f32_16x16x32_f16(
          false, fa.v, false, fb.v, (short)0, acc[nt], false, false);
    }
    __syncthreads();
  }

#pragma unroll
  for (int nt = 0; nt < 2; ++nt) {
    const int n = n0 + nt * 16 + l16;
    const float bv = bias ? bias[n] : 0.f;
    const float sc = bns  ? bns[n]  : 1.f;
    const float so = bnsh ? bnsh[n] : 0.f;
#pragma unroll
    for (int r = 0; r < 8; ++r) {
      long m = m0 + wave * 16 + r + 8 * half;
      float val = acc[nt][r] + bv;
      if (bns) val = fmaxf(val * sc + so, 0.f);
      if (out_f16) ((_Float16*)Out)[m * ldo + n] = (_Float16)val;
      else         ((float*)Out)[m * ldo + n]    = val;
    }
  }
}

// ---------------------------------------------------------------------------
// Attention: logits[v,v'] = sum_{c,t} a[n,c,t,v]*b[n,c,t,v'] / 4800
// softmax over rows (axis=-2), + A + PA.
// Stored TRANSPOSED as f16 [v'][v], zero-padded 32x32 == WMMA B [n][k] layout.
// ---------------------------------------------------------------------------
__global__ __launch_bounds__(640)
void k_att(const _Float16* __restrict__ AB, const float* __restrict__ Aadj,
           const float* __restrict__ PA, _Float16* __restrict__ ATTP) {
  __shared__ float As[400], Bs[400];
  __shared__ float lg[625];
  __shared__ float cmax[25], csum[25];
  const int n = blockIdx.x, i = blockIdx.y;
  const int tid = threadIdx.x;
  const int v = tid / 25, vp = tid % 25;
  float acc = 0.f;
  for (int t = 0; t < 300; ++t) {
    for (int q = tid; q < 800; q += 640) {
      int hs = q / 400, q2 = q % 400;
      int vs = q2 >> 4, c = q2 & 15;
      long base = (((long)(n * 300 + t)) * 25 + vs) * 96 + i * 16 + c;
      float val = (float)AB[base + (hs ? 48 : 0)];
      (hs ? Bs : As)[q2] = val;
    }
    __syncthreads();
    if (tid < 625) {
#pragma unroll
      for (int c = 0; c < 16; ++c) acc += As[v * 16 + c] * Bs[vp * 16 + c];
    }
    __syncthreads();
  }
  if (tid < 625) lg[tid] = acc * (1.f / 4800.f);
  __syncthreads();
  if (tid < 25) {
    float mx = -1e30f;
    for (int r = 0; r < 25; ++r) mx = fmaxf(mx, lg[r * 25 + tid]);
    float sm = 0.f;
    for (int r = 0; r < 25; ++r) sm += expf(lg[r * 25 + tid] - mx);
    cmax[tid] = mx; csum[tid] = sm;
  }
  __syncthreads();
  // destination index q = nn(v')*32 + kk(v): transposed store
  for (int q = tid; q < 1024; q += 640) {
    int nn = q >> 5, kk = q & 31;
    float val = 0.f;
    if (kk < 25 && nn < 25)
      val = expf(lg[kk * 25 + nn] - cmax[nn]) / csum[nn]
          + Aadj[(i * 25 + kk) * 25 + nn] + PA[(i * 25 + kk) * 25 + nn];
    ATTP[(long)(n * 3 + i) * 1024 + q] = (_Float16)val;
  }
}

// ---------------------------------------------------------------------------
// y = sum_i U_i * att_i ; x1 = relu(bn(y + sum bd) + x). WMMA over K=v(pad32).
// Writes x1 (n,c,t,v) f32 and x1 as f16 into CT cols 0..63.
// ---------------------------------------------------------------------------
__global__ __launch_bounds__(128)
void k_gcn_apply(const _Float16* __restrict__ U, const _Float16* __restrict__ ATTP,
                 const float* __restrict__ x, const float* __restrict__ sumbd,
                 const float* __restrict__ bns, const float* __restrict__ bnsh,
                 float* __restrict__ x1, _Float16* __restrict__ CT) {
  __shared__ __align__(16) _Float16 la[64 * LSTR];
  __shared__ __align__(16) _Float16 lb[32 * LSTR];
  const int tid  = threadIdx.x;
  const int wave = tid >> 5, lane = tid & 31;
  const int half = lane >> 4, l16 = lane & 15;
  const int mb = blockIdx.x;    /* 0..299: 64-row block of M=19200 (co*300+t) */
  const int n  = blockIdx.y;    /* batch */
  v8f acc[2] = {v8f{}, v8f{}};

  for (int i = 0; i < 3; ++i) {
    for (int q = tid; q < 64 * 32; q += 128) {
      int row = q >> 5, kk = q & 31;
      int m = mb * 64 + row;
      int co = m / 300, t = m % 300;
      _Float16 val = (_Float16)0.f;
      if (kk < 25)
        val = U[(((long)(n * 300 + t)) * 25 + kk) * 192 + i * 64 + co];
      la[row * LSTR + kk] = val;
    }
    // att^T tile: contiguous [n][k] 16B chunks -> async copy into skewed LDS
    {
      int nn = tid >> 2, kk0 = (tid & 3) << 3;
      copy16_g2l(&lb[nn * LSTR + kk0],
                 ATTP + (long)(n * 3 + i) * 1024 + tid * 8);
    }
    async_wait_all();
    __syncthreads();
    union { uint4 u[2]; v16h v; } fa, fb;
    const int mr = wave * 16 + l16;
    fa.u[0] = *(const uint4*)(&la[mr * LSTR + 8 * half]);
    fa.u[1] = *(const uint4*)(&la[mr * LSTR + 16 + 8 * half]);
#pragma unroll
    for (int nt = 0; nt < 2; ++nt) {
      fb.u[0] = *(const uint4*)(&lb[(nt * 16 + l16) * LSTR + 8 * half]);
      fb.u[1] = *(const uint4*)(&lb[(nt * 16 + l16) * LSTR + 16 + 8 * half]);
      acc[nt] = __builtin_amdgcn_wmma_f32_16x16x32_f16(
          false, fa.v, false, fb.v, (short)0, acc[nt], false, false);
    }
    __syncthreads();
  }

#pragma unroll
  for (int nt = 0; nt < 2; ++nt) {
    int vv = nt * 16 + l16;
    if (vv >= 25) continue;
#pragma unroll
    for (int r = 0; r < 8; ++r) {
      int m = mb * 64 + wave * 16 + r + 8 * half;
      int co = m / 300, t = m % 300;
      float val = acc[nt][r] + sumbd[co];
      val = val * bns[co] + bnsh[co]
          + x[(((long)(n * 64 + co)) * 300 + t) * 25 + vv];
      val = fmaxf(val, 0.f);
      x1[(((long)(n * 64 + co)) * 300 + t) * 25 + vv] = val;
      CT[(((long)(n * 300 + t)) * 25 + vv) * 160 + co] = (_Float16)val;
    }
  }
}

// ---------------------------------------------------------------------------
// pyconv stage (VALU): y1 = conv k=(2R1+1) over 32 ch; y2 grouped (4 groups)
// k=(2R2+1); concat -> bn -> relu; reads/writes f16 columns of CT.
// ---------------------------------------------------------------------------
template<int R1, int R2>
__global__ __launch_bounds__(256)
void k_pyconv(_Float16* __restrict__ CT, int in_off, int out_off,
              const float* __restrict__ W1, const float* __restrict__ b1,
              const float* __restrict__ W2, const float* __restrict__ b2,
              const float* __restrict__ bns, const float* __restrict__ bnsh) {
  constexpr int K1 = 2 * R1 + 1, K2 = 2 * R2 + 1;
  __shared__ float w1[16 * 32 * K1];
  __shared__ float w2[16 * 8 * K2];
  __shared__ float sb1[16], sb2[16], ss[32], ssh[32];
  const int tid = threadIdx.x;
  for (int q = tid; q < 16 * 32 * K1; q += 256) w1[q] = W1[q];
  for (int q = tid; q < 16 * 8 * K2; q += 256) w2[q] = W2[q];
  if (tid < 16) { sb1[tid] = b1[tid]; sb2[tid] = b2[tid]; }
  if (tid < 32) { ss[tid] = bns[tid]; ssh[tid] = bnsh[tid]; }
  __syncthreads();

  long s = (long)blockIdx.x * 256 + tid;
  int n = (int)(s / 7500); int r = (int)(s % 7500); int t = r / 25; int v = r % 25;

  float a1[16], a2[16];
#pragma unroll
  for (int co = 0; co < 16; ++co) { a1[co] = sb1[co]; a2[co] = sb2[co]; }

  for (int dt = -R2; dt <= R2; ++dt) {
    int tt = t + dt;
    if (tt < 0 || tt >= 300) continue;
    union { uint4 u[4]; _Float16 h[32]; } row;
    const uint4* p = (const uint4*)(CT + (((long)(n * 300 + tt)) * 25 + v) * 160 + in_off);
    row.u[0] = p[0]; row.u[1] = p[1]; row.u[2] = p[2]; row.u[3] = p[3];
    float rr[32];
#pragma unroll
    for (int c = 0; c < 32; ++c) rr[c] = (float)row.h[c];
    if (dt >= -R1 && dt <= R1) {
      int kk = dt + R1;
#pragma unroll
      for (int co = 0; co < 16; ++co)
#pragma unroll
        for (int c = 0; c < 32; ++c)
          a1[co] += w1[(co * 32 + c) * K1 + kk] * rr[c];
    }
    int k2 = dt + R2;
#pragma unroll
    for (int co = 0; co < 16; ++co) {
      int g8 = (co >> 2) * 8;
#pragma unroll
      for (int c = 0; c < 8; ++c)
        a2[co] += w2[(co * 8 + c) * K2 + k2] * rr[g8 + c];
    }
  }

  union { uint4 u[4]; _Float16 h[32]; } outp;
#pragma unroll
  for (int co = 0; co < 16; ++co) {
    outp.h[co]      = (_Float16)fmaxf(a1[co] * ss[co]      + ssh[co],      0.f);
    outp.h[16 + co] = (_Float16)fmaxf(a2[co] * ss[16 + co] + ssh[16 + co], 0.f);
  }
  uint4* po = (uint4*)(CT + (((long)(n * 300 + t)) * 25 + v) * 160 + out_off);
  po[0] = outp.u[0]; po[1] = outp.u[1]; po[2] = outp.u[2]; po[3] = outp.u[3];
}

// ---------------------------------------------------------------------------
// DCT channel reduction: yv[n,c] = sum_{t,v} h[(n,t,v),c] * dctw[c,t,v]
// ---------------------------------------------------------------------------
__global__ __launch_bounds__(256)
void k_dct_reduce(const float* __restrict__ h, const float* __restrict__ dctw,
                  float* __restrict__ yv) {
  int n = blockIdx.x, c = blockIdx.y, tid = threadIdx.x;
  float acc = 0.f;
  for (int idx = tid; idx < 7500; idx += 256)
    acc += h[(((long)n * 7500) + idx) * 64 + c] * dctw[c * 7500 + idx];
  __shared__ float red[256];
  red[tid] = acc; __syncthreads();
  for (int s2 = 128; s2 > 0; s2 >>= 1) {
    if (tid < s2) red[tid] += red[tid + s2];
    __syncthreads();
  }
  if (tid == 0) yv[n * 64 + c] = red[0];
}

__global__ void k_fca(const float* __restrict__ yv, const float* __restrict__ F1,
                      const float* __restrict__ F2, float* __restrict__ attc) {
  int n = blockIdx.x, tid = threadIdx.x;   /* 64 threads */
  __shared__ float ys[64], hid[16];
  ys[tid] = yv[n * 64 + tid];
  __syncthreads();
  if (tid < 16) {
    float s = 0.f;
    for (int c = 0; c < 64; ++c) s += ys[c] * F1[tid * 64 + c];
    hid[tid] = fmaxf(s, 0.f);
  }
  __syncthreads();
  float o = 0.f;
  for (int j = 0; j < 16; ++j) o += hid[j] * F2[tid * 16 + j];
  attc[n * 64 + tid] = 1.f / (1.f + expf(-o));
}

__global__ __launch_bounds__(256)
void k_final(const float* __restrict__ h, const float* __restrict__ attc,
             const float* __restrict__ x1, float* __restrict__ out) {
  long idx = (long)blockIdx.x * 256 + threadIdx.x;   /* (n,c,t,v) order */
  int v = (int)(idx % 25); long q = idx / 25;
  int t = (int)(q % 300);  q /= 300;
  int c = (int)(q % 64);   int n = (int)(q / 64);
  float hv = h[(((long)(n * 300 + t)) * 25 + v) * 64 + c];
  out[idx] = fmaxf(hv * attc[n * 64 + c] + x1[idx], 0.f);
}

// ---------------------------------------------------------------------------
// host orchestration
// ---------------------------------------------------------------------------
extern "C" void kernel_launch(void* const* d_in, const int* in_sizes, int n_in,
                              void* d_out, int out_size, void* d_ws, size_t ws_size,
                              hipStream_t stream) {
  (void)in_sizes; (void)n_in; (void)out_size; (void)ws_size;
  // setup_inputs() flattening assumption: dict leaves in order, bn tuples -> 4 arrays
  const float* x    = (const float*)d_in[0];
  const float* Aadj = (const float*)d_in[1];
  const float* PA   = (const float*)d_in[2];
  const float* Wa   = (const float*)d_in[3];
  const float* ba   = (const float*)d_in[4];
  const float* Wb   = (const float*)d_in[5];
  const float* bb   = (const float*)d_in[6];
  const float* Wd   = (const float*)d_in[7];
  const float* bd   = (const float*)d_in[8];
  const float* bnG[4]  = {(const float*)d_in[9],  (const float*)d_in[10], (const float*)d_in[11], (const float*)d_in[12]};
  const float* W0   = (const float*)d_in[13];
  const float* bn0p[4] = {(const float*)d_in[14], (const float*)d_in[15], (const float*)d_in[16], (const float*)d_in[17]};
  const float* W11  = (const float*)d_in[18];
  const float* b11  = (const float*)d_in[19];
  const float* W12  = (const float*)d_in[20];
  const float* b12  = (const float*)d_in[21];
  const float* bn1p[4] = {(const float*)d_in[22], (const float*)d_in[23], (const float*)d_in[24], (const float*)d_in[25]};
  const float* W21  = (const float*)d_in[26];
  const float* b21  = (const float*)d_in[27];
  const float* W22  = (const float*)d_in[28];
  const float* b22  = (const float*)d_in[29];
  const float* bn2p[4] = {(const float*)d_in[30], (const float*)d_in[31], (const float*)d_in[32], (const float*)d_in[33]};
  const float* Wc   = (const float*)d_in[34];
  const float* bncp[4] = {(const float*)d_in[35], (const float*)d_in[36], (const float*)d_in[37], (const float*)d_in[38]};
  const float* F1   = (const float*)d_in[39];
  const float* F2   = (const float*)d_in[40];

  uint8_t* ws = (uint8_t*)d_ws;
  size_t off = 0;
  auto take = [&](size_t bytes) -> uint8_t* {
    uint8_t* p = ws + off;
    off += (bytes + 255) & ~(size_t)255;
    return p;
  };

  float* bnGs  = (float*)take(64 * 4);  float* bnGsh = (float*)take(64 * 4);
  float* bn0s  = (float*)take(32 * 4);  float* bn0sh = (float*)take(32 * 4);
  float* bn1s  = (float*)take(32 * 4);  float* bn1sh = (float*)take(32 * 4);
  float* bn2s  = (float*)take(32 * 4);  float* bn2sh = (float*)take(32 * 4);
  float* bncs  = (float*)take(64 * 4);  float* bncsh = (float*)take(64 * 4);
  _Float16* wab    = (_Float16*)take(64 * 96 * 2);
  float*    biasab = (float*)take(96 * 4);
  _Float16* wdp    = (_Float16*)take(64 * 192 * 2);
  float*    sumbd  = (float*)take(64 * 4);
  _Float16* w0h    = (_Float16*)take(64 * 32 * 2);
  _Float16* wch    = (_Float16*)take(160 * 64 * 2);
  float*    dctw   = (float*)take((size_t)64 * 7500 * 4);
  _Float16* attp   = (_Float16*)take((size_t)192 * 1024 * 2);
  float*    yv     = (float*)take(64 * 64 * 4);
  float*    attc   = (float*)take(64 * 64 * 4);

  const size_t xh_b = (size_t)NTV * 64 * 2;   /* 61.44 MB  */
  const size_t ab_b = (size_t)NTV * 96 * 2;   /* 92.16 MB  */
  const size_t ct_b = (size_t)NTV * 160 * 2;  /* 153.6 MB  */
  const size_t u_b  = (size_t)NTV * 192 * 2;  /* 184.3 MB  */
  // region1: xh+AB (phases 1-3) later reused by CT (phases 4-6)
  uint8_t* region1 = take((xh_b + ab_b) > ct_b ? (xh_b + ab_b) : ct_b);
  _Float16* xh  = (_Float16*)region1;
  _Float16* ab  = (_Float16*)(region1 + xh_b);
  _Float16* ct  = (_Float16*)region1;
  // region2: U (phases 2-4) later reused by h f32 (phases 6-7)
  uint8_t* region2 = take(u_b);
  _Float16* ubuf = (_Float16*)region2;
  float*    hbuf = (float*)region2;
  float*    x1   = (float*)take((size_t)NTV * 64 * 4);
  float*    out  = (float*)d_out;

  // --- prep ---
  k_prep_bn<<<1, 64, 0, stream>>>(bnG[0], bnG[1], bnG[2], bnG[3], bnGs, bnGsh, 64);
  k_prep_bn<<<1, 32, 0, stream>>>(bn0p[0], bn0p[1], bn0p[2], bn0p[3], bn0s, bn0sh, 32);
  k_prep_bn<<<1, 32, 0, stream>>>(bn1p[0], bn1p[1], bn1p[2], bn1p[3], bn1s, bn1sh, 32);
  k_prep_bn<<<1, 32, 0, stream>>>(bn2p[0], bn2p[1], bn2p[2], bn2p[3], bn2s, bn2sh, 32);
  k_prep_bn<<<1, 64, 0, stream>>>(bncp[0], bncp[1], bncp[2], bncp[3], bncs, bncsh, 64);
  k_prep_wab<<<24, 256, 0, stream>>>(Wa, ba, Wb, bb, wab, biasab);
  k_prep_wd<<<48, 256, 0, stream>>>(Wd, bd, wdp, sumbd);
  k_prep_w0<<<8, 256, 0, stream>>>(W0, w0h);
  k_prep_wc<<<40, 256, 0, stream>>>(Wc, wch);
  k_dctw<<<1875, 256, 0, stream>>>(dctw);
  k_xh<<<120000, 256, 0, stream>>>(x, xh);

  // --- GCN: a|b projections (K=64, N=96), +bias, f16 out ---
  k_gemm<<<dim3(7500, 3), 128, 0, stream>>>(xh, 64, wab, 96, 64,
                                            (void*)ab, 96, 1, biasab, nullptr, nullptr);
  // --- attention matrices (softmax axis=-2, +A+PA), transposed padded f16 ---
  k_att<<<dim3(64, 3), 640, 0, stream>>>(ab, Aadj, PA, attp);
  // --- U_i = Wd_i * x for all subsets (K=64, N=192), f16 out ---
  k_gemm<<<dim3(7500, 6), 128, 0, stream>>>(xh, 64, wdp, 192, 64,
                                            (void*)ubuf, 192, 1, nullptr, nullptr, nullptr);
  // --- x1 = relu(bn(sum_i U_i*att_i + sum bd) + x); also f16 into CT[:,0:64] ---
  k_gcn_apply<<<dim3(300, 64), 128, 0, stream>>>(ubuf, attp, x, sumbd, bnGs, bnGsh, x1, ct);

  // --- TCN: W0 1x1 (K=64, N=32), bn0+relu -> CT[:,64:96] ---
  k_gemm<<<dim3(7500, 1), 128, 0, stream>>>(ct, 160, w0h, 32, 64,
                                            (void*)(ct + 64), 160, 1, nullptr, bn0s, bn0sh);
  // --- pyconv stages ---
  k_pyconv<1, 2><<<1875, 256, 0, stream>>>(ct, 64, 96,  W11, b11, W12, b12, bn1s, bn1sh);
  k_pyconv<3, 4><<<1875, 256, 0, stream>>>(ct, 96, 128, W21, b21, W22, b22, bn2s, bn2sh);
  // --- Wc 1x1 on concat (K=160, N=64), bnc+relu, f32 out ---
  k_gemm<<<dim3(7500, 2), 128, 0, stream>>>(ct, 160, wch, 64, 160,
                                            (void*)hbuf, 64, 0, nullptr, bncs, bncsh);
  // --- FcaNet attention + final residual relu ---
  k_dct_reduce<<<dim3(64, 64), 256, 0, stream>>>(hbuf, dctw, yv);
  k_fca<<<64, 64, 0, stream>>>(yv, F1, F2, attc);
  k_final<<<120000, 256, 0, stream>>>(hbuf, attc, x1, out);
}
